// PredRNN_43774306681285
// MI455X (gfx1250) — matrix operand order repaired and checked by reading, not compiled, not run on hardware
//
#include <hip/hip_runtime.h>

typedef unsigned short u16;
typedef __attribute__((ext_vector_type(16))) __bf16 v16bf;
typedef __attribute__((ext_vector_type(8)))  float  v8f;

#define NP 16384   // B * 64 * 64 pixels
#define EPS 1e-3f

// ---------------------------------------------------------------- helpers
__device__ __forceinline__ u16 f2bf(float f) {
    unsigned u = __float_as_uint(f);
    unsigned r = (u + 0x7FFFu + ((u >> 16) & 1u)) >> 16;
    return (u16)r;
}
__device__ __forceinline__ unsigned pack2bf(float a, float b) {
    return (unsigned)f2bf(a) | ((unsigned)f2bf(b) << 16);
}
__device__ __forceinline__ float sigm(float x) { return 1.0f / (1.0f + __expf(-x)); }

// LN stats from a reduction slot: returns mean, sets istd
__device__ __forceinline__ float ln_stat(const float* __restrict__ r, int b, float inv, float& istd) {
    float m = r[b * 2] * inv;
    float v = r[b * 2 + 1] * inv - m * m;
    istd = rsqrtf(v + EPS);
    return m;
}

// ---------------------------------------------------------------- small kernels
__global__ void pack_bf16_k(const float* __restrict__ s, u16* __restrict__ d, int n) {
    int i = blockIdx.x * 256 + threadIdx.x;
    if (i < n) d[i] = f2bf(s[i]);
}
__global__ void zero_k(float* p, int n) {
    int i = blockIdx.x * 256 + threadIdx.x;
    if (i < n) p[i] = 0.0f;
}

// ---------------------------------------------------------------- WMMA implicit-GEMM conv
// Reads fp32 NHWC activations (Xa/Xb channel-concat, CinA channels from Xa), converts to
// bf16 on the fly into LDS, multiplies against pre-packed bf16 weights Wt [KS*KS*Cin, Cout],
// writes fp32 Y and (optionally) accumulates per-batch sum/sumsq into `red` for fused TLN.
// Block: 256 thr (8 waves). Tile: 128 M x 64 N. Wave: 2x2 WMMA f32_16x16x32_bf16.
template <int KSIZE>
__global__ __launch_bounds__(256) void conv_wmma_k(
    const float* __restrict__ Xa, const float* __restrict__ Xb, int CinA,
    const u16* __restrict__ Wt, const float* __restrict__ bias,
    float* __restrict__ Y, float* __restrict__ red,
    int Cin, int Cout)
{
    constexpr int PAD = KSIZE / 2;
    __shared__ u16 As[2][128][32];   // [m-row][k]
    __shared__ u16 Bs[2][64][32];    // [n][k]

    const int tid   = threadIdx.x;
    const int lane  = tid & 31;
    const int wave  = tid >> 5;
    const int waveM = (wave >> 1) << 5;
    const int waveN = (wave & 1) << 5;
    const int mBlock = blockIdx.x << 7;
    const int nBlock = blockIdx.y << 6;
    const int Ktot = KSIZE * KSIZE * Cin;
    const int nk   = Ktot >> 5;
    const int batch = mBlock >> 12;          // whole block lies in one batch
    const int CinB = Cin - CinA;

    // A-loader geometry: thread -> (row, half of 32-K chunk => 16 channels)
    const int arow  = tid >> 1;
    const int ahalf = tid & 1;
    const int pm  = mBlock + arow;
    const int ab  = pm >> 12;
    const int ay  = (pm & 4095) >> 6;
    const int ax  = pm & 63;
    // B-loader geometry
    const int bkr = tid >> 3;
    const int bnb = (tid & 7) << 3;

    auto loadChunk = [&](int kc, int buf) {
        const int k0 = kc << 5;
        // ---- A tile (im2col, fp32 -> bf16): 32 K values lie inside one tap (Cin%32==0)
        int tap = k0 / Cin;
        int ci  = k0 - tap * Cin + (ahalf << 4);
        int ky  = tap / KSIZE;
        int kx  = tap - ky * KSIZE;
        int sy  = ay + ky - PAD;
        int sx  = ax + kx - PAD;
        uint4 v0 = make_uint4(0u, 0u, 0u, 0u), v1 = v0;
        if (sy >= 0 && sy < 64 && sx >= 0 && sx < 64) {
            const int pix = (ab << 12) + (sy << 6) + sx;
            const float4* src;
            if (ci < CinA)
                src = reinterpret_cast<const float4*>(Xa + (size_t)pix * CinA + ci);
            else
                src = reinterpret_cast<const float4*>(Xb + (size_t)pix * CinB + (ci - CinA));
            float4 q0 = src[0], q1 = src[1], q2 = src[2], q3 = src[3];
            v0 = make_uint4(pack2bf(q0.x, q0.y), pack2bf(q0.z, q0.w),
                            pack2bf(q1.x, q1.y), pack2bf(q1.z, q1.w));
            v1 = make_uint4(pack2bf(q2.x, q2.y), pack2bf(q2.z, q2.w),
                            pack2bf(q3.x, q3.y), pack2bf(q3.z, q3.w));
        }
        *reinterpret_cast<uint4*>(&As[buf][arow][(ahalf << 4) + 0]) = v0;
        *reinterpret_cast<uint4*>(&As[buf][arow][(ahalf << 4) + 8]) = v1;
        // ---- B tile: coalesced read, transposed store to [n][k]
        uint4 w = *reinterpret_cast<const uint4*>(
            Wt + (size_t)(k0 + bkr) * Cout + nBlock + bnb);
        const u16* wp = reinterpret_cast<const u16*>(&w);
#pragma unroll
        for (int j = 0; j < 8; ++j) Bs[buf][bnb + j][bkr] = wp[j];
        if (k0 + 32 < Ktot)   // prefetch next weight chunk (global_prefetch_b8)
            __builtin_prefetch(Wt + (size_t)(k0 + 32 + bkr) * Cout + nBlock + bnb, 0, 1);
    };

    const v8f vz = {0.f, 0.f, 0.f, 0.f, 0.f, 0.f, 0.f, 0.f};
    v8f acc[2][2] = {{vz, vz}, {vz, vz}};

    loadChunk(0, 0);
    __syncthreads();

    const int mrow = lane & 15;
    const int half = lane >> 4;

    for (int kc = 0; kc < nk; ++kc) {
        const int buf = kc & 1;
        if (kc + 1 < nk) loadChunk(kc + 1, buf ^ 1);

        union { uint4 q[2]; v16bf v; } aF[2], bF[2];
#pragma unroll
        for (int i = 0; i < 2; ++i) {
            aF[i].q[0] = *reinterpret_cast<const uint4*>(&As[buf][waveM + (i << 4) + mrow][(half << 3) + 0]);
            aF[i].q[1] = *reinterpret_cast<const uint4*>(&As[buf][waveM + (i << 4) + mrow][16 + (half << 3)]);
            bF[i].q[0] = *reinterpret_cast<const uint4*>(&Bs[buf][waveN + (i << 4) + mrow][(half << 3) + 0]);
            bF[i].q[1] = *reinterpret_cast<const uint4*>(&Bs[buf][waveN + (i << 4) + mrow][16 + (half << 3)]);
        }
#pragma unroll
        for (int i = 0; i < 2; ++i)
#pragma unroll
            for (int j = 0; j < 2; ++j)
                acc[i][j] = __builtin_amdgcn_wmma_f32_16x16x32_bf16(
                    false, aF[i].v, false, bF[j].v, (short)0, acc[i][j], false, false);
        __syncthreads();
    }

    // Epilogue: write Y = acc + bias, fused partial sum/sumsq for tensor-layer-norm
    float s1 = 0.f, s2 = 0.f;
#pragma unroll
    for (int i = 0; i < 2; ++i)
#pragma unroll
        for (int j = 0; j < 2; ++j)
#pragma unroll
            for (int r = 0; r < 8; ++r) {
                int row = mBlock + waveM + (i << 4) + (half << 3) + r;
                int col = nBlock + waveN + (j << 4) + mrow;
                float v = acc[i][j][r] + bias[col];
                Y[(size_t)row * Cout + col] = v;
                s1 += v; s2 += v * v;
            }
    if (red) {
#pragma unroll
        for (int o = 16; o > 0; o >>= 1) {
            s1 += __shfl_down(s1, o, 32);
            s2 += __shfl_down(s2, o, 32);
        }
        if (lane == 0) {
            atomicAdd(&red[batch * 2 + 0], s1);
            atomicAdd(&red[batch * 2 + 1], s2);
        }
    }
}

// ---------------------------------------------------------------- direct conv, Cin = 1 (lstm0 x-conv), fused reduce
__global__ void conv5_cin1_k(const float* __restrict__ X, const float* __restrict__ W,
                             const float* __restrict__ bias, float* __restrict__ Y,
                             float* __restrict__ red, int Cout)
{
    __shared__ float taps[25];
    __shared__ float w1[4], w2[4];
    int pmv = blockIdx.x;
    int b = pmv >> 12, pix = pmv & 4095, y = pix >> 6, x = pix & 63;
    if (threadIdx.x < 25) {
        int ky = threadIdx.x / 5, kx = threadIdx.x % 5;
        int sy = y + ky - 2, sx = x + kx - 2;
        taps[threadIdx.x] = (sy >= 0 && sy < 64 && sx >= 0 && sx < 64)
                                ? X[(b << 12) + (sy << 6) + sx] : 0.0f;
    }
    __syncthreads();
    float s1 = 0.f, s2 = 0.f;
    for (int co = threadIdx.x; co < Cout; co += blockDim.x) {
        float s = bias[co];
#pragma unroll
        for (int t = 0; t < 25; ++t) s += taps[t] * W[t * Cout + co];
        Y[(size_t)pmv * Cout + co] = s;
        s1 += s; s2 += s * s;
    }
#pragma unroll
    for (int o = 16; o > 0; o >>= 1) {
        s1 += __shfl_down(s1, o, 32);
        s2 += __shfl_down(s2, o, 32);
    }
    int wv = threadIdx.x >> 5, ln = threadIdx.x & 31;
    if (ln == 0) { w1[wv] = s1; w2[wv] = s2; }
    __syncthreads();
    if (threadIdx.x == 0) {
        atomicAdd(&red[b * 2 + 0], w1[0] + w1[1] + w1[2] + w1[3]);
        atomicAdd(&red[b * 2 + 1], w2[0] + w2[1] + w2[2] + w2[3]);
    }
}

// ---------------------------------------------------------------- gate kernels (LN fused)
// x_cc split: [i_x,g_x,f_x,o_x,i_x2,g_x2,f_x2]; h_cc: [i,g,f,o]; c_cc/m_cc: 3 chunks; c2m: [i,g,f,o]
#define INV_X  (1.0f / 1835008.0f)   // 4096*448 per batch
#define INV_H  (1.0f / 1048576.0f)   // 4096*256
#define INV_C3 (1.0f / 786432.0f)    // 4096*192
#define INV_O  (1.0f / 262144.0f)    // 4096*64
#define INV_G  (1.0f / 524288.0f)    // 4096*128

__global__ void gates1_k(const float* __restrict__ xcc, const float* __restrict__ rx,
                         const float* __restrict__ gxv, const float* __restrict__ bxv,
                         const float* __restrict__ hcc, const float* __restrict__ rh,
                         const float* __restrict__ ghv, const float* __restrict__ bhv,
                         const float* __restrict__ ccc, const float* __restrict__ rc,
                         const float* __restrict__ gcv, const float* __restrict__ bcv,
                         float* __restrict__ c)
{
    int i = blockIdx.x * 256 + threadIdx.x;
    if (i >= NP * 64) return;
    int p = i >> 6, ch = i & 63, b = p >> 12;
    float ix, mx = ln_stat(rx, b, INV_X, ix);
    float ih, mh = ln_stat(rh, b, INV_H, ih);
    float ic, mc = ln_stat(rc, b, INV_C3, ic);
    const float* xp = xcc + (size_t)p * 448;
    const float* hp = hcc + (size_t)p * 256;
    const float* cp = ccc + (size_t)p * 192;
    auto LX = [&](int cc) { return (xp[cc] - mx) * ix * gxv[cc] + bxv[cc]; };
    auto LH = [&](int cc) { return (hp[cc] - mh) * ih * ghv[cc] + bhv[cc]; };
    auto LC = [&](int cc) { return (cp[cc] - mc) * ic * gcv[cc] + bcv[cc]; };
    float I = sigm(LX(ch)        + LH(ch)        + LC(ch));
    float G = tanhf(LX(64 + ch)  + LH(64 + ch)   + LC(64 + ch));
    float F = sigm(LX(128 + ch)  + LH(128 + ch)  + LC(128 + ch) + 1.0f);
    c[i] = F * c[i] + I * G;
}

__global__ void gates2_k(const float* __restrict__ c2m, const float* __restrict__ rq,
                         const float* __restrict__ gqv, const float* __restrict__ bqv,
                         const float* __restrict__ xcc, const float* __restrict__ rx,
                         const float* __restrict__ gxv, const float* __restrict__ bxv,
                         const float* __restrict__ mcc, const float* __restrict__ rm,
                         const float* __restrict__ gmv, const float* __restrict__ bmv,
                         float* __restrict__ m)
{
    int i = blockIdx.x * 256 + threadIdx.x;
    if (i >= NP * 64) return;
    int p = i >> 6, ch = i & 63, b = p >> 12;
    float iq, mq = ln_stat(rq, b, INV_H, iq);
    float ix, mx = ln_stat(rx, b, INV_X, ix);
    float im, mm = ln_stat(rm, b, INV_C3, im);
    const float* qp = c2m + (size_t)p * 256;
    const float* xp = xcc + (size_t)p * 448;
    const float* mp = mcc + (size_t)p * 192;
    auto LQ = [&](int cc) { return (qp[cc] - mq) * iq * gqv[cc] + bqv[cc]; };
    auto LX = [&](int cc) { return (xp[cc] - mx) * ix * gxv[cc] + bxv[cc]; };
    auto LM = [&](int cc) { return (mp[cc] - mm) * im * gmv[cc] + bmv[cc]; };
    float ii = sigm(LQ(ch)        + LX(256 + ch) + LM(ch));
    float gg = tanhf(LQ(64 + ch)  + LX(320 + ch));
    float ff = sigm(LQ(128 + ch)  + LX(384 + ch) + LM(64 + ch) + 1.0f);
    m[i] = ff * tanhf(LM(128 + ch)) + ii * gg;
}

__global__ void gates3_k(const float* __restrict__ xcc, const float* __restrict__ rx,
                         const float* __restrict__ gxv, const float* __restrict__ bxv,
                         const float* __restrict__ hcc, const float* __restrict__ rh,
                         const float* __restrict__ ghv, const float* __restrict__ bhv,
                         const float* __restrict__ c2m, const float* __restrict__ rq,
                         const float* __restrict__ gqv, const float* __restrict__ bqv,
                         const float* __restrict__ om,  const float* __restrict__ ro,
                         const float* __restrict__ gov, const float* __restrict__ bov,
                         const float* __restrict__ cell, float* __restrict__ h)
{
    int i = blockIdx.x * 256 + threadIdx.x;
    if (i >= NP * 64) return;
    int p = i >> 6, ch = i & 63, b = p >> 12;
    float ix, mx = ln_stat(rx, b, INV_X, ix);
    float ih, mh = ln_stat(rh, b, INV_H, ih);
    float iq, mq = ln_stat(rq, b, INV_H, iq);
    float io, mo = ln_stat(ro, b, INV_O, io);
    float vx = (xcc[(size_t)p * 448 + 192 + ch] - mx) * ix * gxv[192 + ch] + bxv[192 + ch];
    float vh = (hcc[(size_t)p * 256 + 192 + ch] - mh) * ih * ghv[192 + ch] + bhv[192 + ch];
    float vq = (c2m[(size_t)p * 256 + 192 + ch] - mq) * iq * gqv[192 + ch] + bqv[192 + ch];
    float vo = (om[i] - mo) * io * gov[ch] + bov[ch];
    float o = tanhf(vx + vh + vq + vo);
    h[i] = o * tanhf(cell[i]);
}

__global__ void ghu_gate_k(const float* __restrict__ zcc, const float* __restrict__ rz,
                           const float* __restrict__ gzv, const float* __restrict__ bzv,
                           const float* __restrict__ xcc2, const float* __restrict__ rxx,
                           const float* __restrict__ gxxv, const float* __restrict__ bxxv,
                           float* __restrict__ z)
{
    int i = blockIdx.x * 256 + threadIdx.x;
    if (i >= NP * 64) return;
    int p = i >> 6, ch = i & 63, b = p >> 12;
    float iz, mz = ln_stat(rz, b, INV_G, iz);
    float ix, mx = ln_stat(rxx, b, INV_G, ix);
    const float* zp = zcc  + (size_t)p * 128;
    const float* xp = xcc2 + (size_t)p * 128;
    auto LZ = [&](int cc) { return (zp[cc] - mz) * iz * gzv[cc] + bzv[cc]; };
    auto LG = [&](int cc) { return (xp[cc] - mx) * ix * gxxv[cc] + bxxv[cc]; };
    float pp = tanhf(LG(ch) + LZ(ch));
    float u  = sigm(LG(64 + ch) + LZ(64 + ch));
    z[i] = u * pp + (1.0f - u) * z[i];
}

// ---------------------------------------------------------------- input select & pixel output
__global__ void select_in_k(const float* __restrict__ images, const float* __restrict__ mask_true,
                            const float* __restrict__ xg, float* __restrict__ inp, int t)
{
    int i = blockIdx.x * 256 + threadIdx.x;
    if (i >= NP) return;
    int b = i >> 12, pix = i & 4095;
    float img = images[((size_t)b * 16 + t) * 4096 + pix];
    float v;
    if (t < 12) v = img;
    else {
        float mk = mask_true[((size_t)b * 3 + (t - 12)) * 4096 + pix];
        v = mk * img + (1.0f - mk) * xg[i];
    }
    inp[i] = v;
}

__global__ void pix_out_k(const float* __restrict__ h1, const float* __restrict__ w,
                          const float* __restrict__ bb, float* __restrict__ xg,
                          float* __restrict__ out, int t)
{
    int p = blockIdx.x * 256 + threadIdx.x;
    if (p >= NP) return;
    float s = bb[0];
    const float* hp = h1 + (size_t)p * 64;
#pragma unroll 8
    for (int c = 0; c < 64; ++c) s += hp[c] * w[c];
    xg[p] = s;
    if (t >= 11) {
        int b = p >> 12, pix = p & 4095;
        out[((size_t)b * 4 + (t - 11)) * 4096 + pix] = s;
    }
}

// ---------------------------------------------------------------- host
struct LstmP {   // sorted-key (pytree) order within each lstm dict
    const float *bc, *bc2m, *bh, *blast, *bm, *bom, *bx,
        *lnc2m_b, *lnc2m_g, *lnc_b, *lnc_g, *lnh_b, *lnh_g, *lnm_b, *lnm_g,
        *lnom_b, *lnom_g, *lnx_b, *lnx_g,
        *wc, *wc2m, *wh, *wlast, *wm, *wom, *wx;
};
struct GhuP { const float *bx, *bz, *lnx_b, *lnx_g, *lnz_b, *lnz_g, *wx, *wz; };
struct WB { u16 *wh, *wc, *wm, *wc2m, *wom, *wlast, *wx; };

extern "C" void kernel_launch(void* const* d_in, const int* in_sizes, int n_in,
                              void* d_out, int out_size, void* d_ws, size_t ws_size,
                              hipStream_t stream)
{
    (void)in_sizes; (void)n_in; (void)out_size; (void)ws_size;

    // -------- inputs (pytree sorted-dict flatten order)
    const float* images    = (const float*)d_in[0];
    const float* mask_true = (const float*)d_in[1];
    const float* bpix      = (const float*)d_in[2];
    GhuP G;
    { const float** f = reinterpret_cast<const float**>(&G);
      for (int i = 0; i < 8; ++i) f[i] = (const float*)d_in[3 + i]; }
    LstmP P0, P1;
    { const float** f = reinterpret_cast<const float**>(&P0);
      for (int i = 0; i < 26; ++i) f[i] = (const float*)d_in[11 + i]; }
    { const float** f = reinterpret_cast<const float**>(&P1);
      for (int i = 0; i < 26; ++i) f[i] = (const float*)d_in[37 + i]; }
    const float* wpix = (const float*)d_in[63];

    // -------- workspace bump allocator (256B aligned)
    char* wsp = (char*)d_ws;
    size_t off = 0;
    auto A = [&](size_t bytes) -> void* {
        void* p = wsp + off;
        off = (off + bytes + 255) & ~(size_t)255;
        return p;
    };
    const int SZ_WH = 409600, SZ_WC = 307200, SZ_WM = 307200, SZ_WC2M = 409600,
              SZ_WOM = 102400, SZ_WLAST = 8192, SZ_WX1 = 716800, SZ_WG = 204800;

    WB W0, W1;
    W0.wh = (u16*)A(SZ_WH * 2);  W0.wc = (u16*)A(SZ_WC * 2);  W0.wm = (u16*)A(SZ_WM * 2);
    W0.wc2m = (u16*)A(SZ_WC2M * 2); W0.wom = (u16*)A(SZ_WOM * 2); W0.wlast = (u16*)A(SZ_WLAST * 2);
    W0.wx = nullptr;
    W1.wh = (u16*)A(SZ_WH * 2);  W1.wc = (u16*)A(SZ_WC * 2);  W1.wm = (u16*)A(SZ_WM * 2);
    W1.wc2m = (u16*)A(SZ_WC2M * 2); W1.wom = (u16*)A(SZ_WOM * 2); W1.wlast = (u16*)A(SZ_WLAST * 2);
    W1.wx = (u16*)A(SZ_WX1 * 2);
    u16* wbz  = (u16*)A(SZ_WG * 2);
    u16* wbgx = (u16*)A(SZ_WG * 2);

    const int NSTATE = 6 * NP * 64 + 2 * NP;
    float* stateBase = (float*)A((size_t)NSTATE * 4);
    float* h0  = stateBase;
    float* c0  = h0  + NP * 64;
    float* h1  = c0  + NP * 64;
    float* c1  = h1  + NP * 64;
    float* mem = c1  + NP * 64;
    float* z   = mem + NP * 64;
    float* xg  = z   + NP * 64;
    float* inp = xg  + NP;

    float* xcc   = (float*)A((size_t)NP * 448 * 4);
    float* hcc   = (float*)A((size_t)NP * 256 * 4);
    float* ccc   = (float*)A((size_t)NP * 192 * 4);
    float* mcc   = (float*)A((size_t)NP * 192 * 4);
    float* c2m   = (float*)A((size_t)NP * 256 * 4);
    float* om    = (float*)A((size_t)NP * 64 * 4);
    float* cellb = (float*)A((size_t)NP * 64 * 4);
    float* gzb   = (float*)A((size_t)NP * 128 * 4);
    float* gxb   = (float*)A((size_t)NP * 128 * 4);
    float* red   = (float*)A(64 * 4);   // 8 slots x [B*2]

    auto RED = [&](int s) { return red + s * 8; };

    // -------- launch helpers
    auto packW = [&](const float* s, u16* d, int n) {
        pack_bf16_k<<<(n + 255) / 256, 256, 0, stream>>>(s, d, n);
    };
    auto conv5 = [&](const float* x, const u16* w, const float* bias, float* y,
                     int Cout, float* rs) {
        dim3 g(NP / 128, Cout / 64);
        conv_wmma_k<5><<<g, 256, 0, stream>>>(x, x, 64, w, bias, y, rs, 64, Cout);
    };
    auto convcat = [&](const float* xa, const float* xb, const u16* w, const float* bias,
                       float* y) {
        dim3 g(NP / 128, 1);
        conv_wmma_k<1><<<g, 256, 0, stream>>>(xa, xb, 64, w, bias, y, nullptr, 128, 64);
    };
    const int NG = (NP * 64 + 255) / 256;

    // -------- bf16 weight conversion (once per call, deterministic)
    packW(P0.wh, W0.wh, SZ_WH);   packW(P0.wc, W0.wc, SZ_WC);   packW(P0.wm, W0.wm, SZ_WM);
    packW(P0.wc2m, W0.wc2m, SZ_WC2M); packW(P0.wom, W0.wom, SZ_WOM); packW(P0.wlast, W0.wlast, SZ_WLAST);
    packW(P1.wh, W1.wh, SZ_WH);   packW(P1.wc, W1.wc, SZ_WC);   packW(P1.wm, W1.wm, SZ_WM);
    packW(P1.wc2m, W1.wc2m, SZ_WC2M); packW(P1.wom, W1.wom, SZ_WOM); packW(P1.wlast, W1.wlast, SZ_WLAST);
    packW(P1.wx, W1.wx, SZ_WX1);
    packW(G.wz, wbz, SZ_WG);  packW(G.wx, wbgx, SZ_WG);

    // -------- zero initial state
    zero_k<<<(NSTATE + 255) / 256, 256, 0, stream>>>(stateBase, NSTATE);

    auto cslstm = [&](const LstmP& P, const WB& W, bool xCin1, const float* xin,
                      float* h, float* c, float* m) {
        zero_k<<<1, 256, 0, stream>>>(red, 64);
        conv5(h, W.wh, P.bh, hcc, 256, RED(0));
        conv5(c, W.wc, P.bc, ccc, 192, RED(1));
        conv5(m, W.wm, P.bm, mcc, 192, RED(2));
        if (xCin1) {
            conv5_cin1_k<<<NP, 128, 0, stream>>>(xin, P.wx, P.bx, xcc, RED(3), 448);
        } else {
            conv5(xin, W.wx, P.bx, xcc, 448, RED(3));
        }
        gates1_k<<<NG, 256, 0, stream>>>(xcc, RED(3), P.lnx_g, P.lnx_b,
                                         hcc, RED(0), P.lnh_g, P.lnh_b,
                                         ccc, RED(1), P.lnc_g, P.lnc_b, c);   // c := c_new
        conv5(c, W.wc2m, P.bc2m, c2m, 256, RED(4));
        gates2_k<<<NG, 256, 0, stream>>>(c2m, RED(4), P.lnc2m_g, P.lnc2m_b,
                                         xcc, RED(3), P.lnx_g, P.lnx_b,
                                         mcc, RED(2), P.lnm_g, P.lnm_b, m);   // m := m_new
        conv5(m, W.wom, P.bom, om, 64, RED(5));
        convcat(c, m, W.wlast, P.blast, cellb);
        gates3_k<<<NG, 256, 0, stream>>>(xcc, RED(3), P.lnx_g, P.lnx_b,
                                         hcc, RED(0), P.lnh_g, P.lnh_b,
                                         c2m, RED(4), P.lnc2m_g, P.lnc2m_b,
                                         om,  RED(5), P.lnom_g, P.lnom_b,
                                         cellb, h);                            // h := h_new
    };

    // -------- 15 recurrent steps
    for (int t = 0; t < 15; ++t) {
        select_in_k<<<(NP + 255) / 256, 256, 0, stream>>>(images, mask_true, xg, inp, t);
        cslstm(P0, W0, true, inp, h0, c0, mem);
        // GHU: z = ghu(h0, z)
        zero_k<<<1, 256, 0, stream>>>(red, 64);
        conv5(z,  wbz,  G.bz, gzb, 128, RED(6));
        conv5(h0, wbgx, G.bx, gxb, 128, RED(7));
        ghu_gate_k<<<NG, 256, 0, stream>>>(gzb, RED(6), G.lnz_g, G.lnz_b,
                                           gxb, RED(7), G.lnx_g, G.lnx_b, z);
        cslstm(P1, W1, false, z, h1, c1, mem);
        pix_out_k<<<(NP + 255) / 256, 256, 0, stream>>>(h1, wpix, bpix, xg, (float*)d_out, t);
    }
}
